// XLSTMDepthCell_67877663146032
// MI455X (gfx1250) — matrix operand order repaired and checked by reading, not compiled, not run on hardware
//
#include <hip/hip_runtime.h>
#include <stdint.h>

// ---------------------------------------------------------------------------
// xLSTM depth-cell stack for MI455X (gfx1250, wave32, WMMA bf16 16x16x32)
// N = 2048 independent tokens, H = 512, NH = 8, DHQK = 32, DHV = 64, F = 1408
// GEMM: block = 8 waves = 128 rows x 64 cols. B staged in LDS by
// global_load_async_to_lds_b128 (ASYNCcnt), 3-deep stage rotation (stage =
// 2 k-tiles = 8 WMMAs per barrier), branch-free with clamped dummy issues.
// ---------------------------------------------------------------------------

typedef __attribute__((ext_vector_type(16))) __bf16 v16bf;
typedef __attribute__((ext_vector_type(8)))  float  v8f;

struct U8 { uint4 lo, hi; };   // 32 bytes == one v16bf fragment

static __device__ __forceinline__ unsigned short f32_to_bf16(float f) {
  union { float f; unsigned u; } v; v.f = f;
  unsigned r = v.u + 0x7FFFu + ((v.u >> 16) & 1u);   // round-to-nearest-even
  return (unsigned short)(r >> 16);
}

// ---------------------------------------------------------------------------
// Repack fp32 weight [K x M] (row-major) into bf16 WMMA B-fragment layout:
//   dst dword index = ((mt*KT + kt)*32 + lane)*8 + v
//   dword v of lane holds bf16 pair (K = kt*32 + (lane>=16 ? 16:0) + 2v, K+1)
//   for column  col = mt*16 + (lane & 15).  col >= M -> zero pad.
// ---------------------------------------------------------------------------
__global__ __launch_bounds__(256) void repack_b_kernel(
    const float* __restrict__ W, unsigned int* __restrict__ dst,
    int K, int M, int tiles, float scale)
{
  const int KT = K >> 5;
  const int total = tiles * KT * 256;
  int t = blockIdx.x * 256 + threadIdx.x;
  if (t >= total) return;
  int mtl  = t / (KT << 8);
  int rem  = t - mtl * (KT << 8);
  int kt   = rem >> 8;
  int r2   = rem & 255;
  int lane = r2 >> 3;
  int v    = r2 & 7;
  int col  = (mtl << 4) + (lane & 15);
  int k0   = (kt << 5) + ((lane >> 4) << 4) + (v << 1);
  float f0 = 0.f, f1 = 0.f;
  if (col < M) {
    f0 = W[(size_t)k0 * M + col] * scale;
    f1 = W[(size_t)(k0 + 1) * M + col] * scale;
  }
  dst[t] = (unsigned)f32_to_bf16(f0) | ((unsigned)f32_to_bf16(f1) << 16);
}

// ---------------------------------------------------------------------------
// GEMM: C[N x M] (+= Cres) = A[N x K](bf16,row-major) * Bpacked, M = Mtiles*16
// Requires: N % 128 == 0, Mtiles % 4 == 0, (K/32) even and >= 4.
// ---------------------------------------------------------------------------
__global__ __launch_bounds__(256) void gemm_bf16_wmma(
    const unsigned short* __restrict__ A,
    const unsigned int*  __restrict__ Bp,
    float* __restrict__ C,
    const float* __restrict__ Cres,
    int N, int K, int Mtiles)
{
  // 3 stage buffers (8KB = 2 k-tiles) + 1 dummy sink buffer
  __shared__ alignas(16) unsigned int smemB[4][2048];
  const int lane = threadIdx.x & 31;
  const int w    = threadIdx.x >> 5;                    // wave 0..7
  const int colGroups = Mtiles >> 2;
  const int blkRow = blockIdx.x / colGroups;
  const int cg     = blockIdx.x - blkRow * colGroups;
  const int row0   = (blkRow * 8 + w) << 4;             // this wave's 16 rows
  const int mt0    = cg << 2;
  const int KT     = K >> 5;
  const int S      = KT >> 1;                           // stages of 2 k-tiles
  const int hi     = lane >> 4;
  const int Mstride = Mtiles << 4;

  // --- async B staging: wave w copies half 'half' of col tile 'jmine' ---
  const int jmine = w >> 1, half = w & 1;
  const char* gb = (const char*)(Bp + (size_t)(mt0 + jmine) * KT * 256)
                   + half * 512 + lane * 16;            // +1024 bytes per k-tile
  const unsigned lbase  = (unsigned)(uintptr_t)(&smemB[0][0])
                        + (unsigned)(jmine * 1024 + half * 512 + lane * 16);
  const unsigned ldummy = lbase + 3u * 8192u;           // sink buffer

  auto issueStage = [&](int gs, unsigned labuf) {       // 2 async b128 ops
    unsigned long long ga0 =
        (unsigned long long)(uintptr_t)gb + (unsigned long long)(2 * gs) * 1024ull;
    asm volatile("global_load_async_to_lds_b128 %0, %1, off"
                 :: "v"(labuf), "v"(ga0) : "memory");
    unsigned long long ga1 = ga0 + 1024ull;
    unsigned la1 = labuf + 4096u;
    asm volatile("global_load_async_to_lds_b128 %0, %1, off"
                 :: "v"(la1), "v"(ga1) : "memory");
  };

  // prologue: stages 0 and 1 in flight (4 ops outstanding)
  issueStage(0, lbase);
  issueStage(1, lbase + 8192u);

  // --- A fragments (contiguous k order, same as packed B) ---
  //  lanes 0-15 : row = lane,    K = kb+0..15
  //  lanes 16-31: row = lane-16, K = kb+16..31
  const unsigned short* arow = A + (size_t)(row0 + (lane & 15)) * K;
  const int aoff = hi << 4;
  auto loadA = [&](int kt) {
    U8 r;
    const int kb = kt << 5;
    r.lo = *(const uint4*)(arow + kb + aoff);
    r.hi = *(const uint4*)(arow + kb + 8 + aoff);
    return r;
  };
  U8 a0 = loadA(0);
  U8 a1 = loadA(1);

  v8f zero = {0.f,0.f,0.f,0.f,0.f,0.f,0.f,0.f};
  v8f acc[4];
#pragma unroll
  for (int j = 0; j < 4; ++j) acc[j] = zero;

  const int ktMax = KT - 1;
  for (int s = 0; s < S; ++s) {
    // prefetch next stage's A fragments (clamped; redundant at tail, harmless)
    int kn0 = 2 * s + 2; kn0 = kn0 > ktMax ? ktMax : kn0;
    int kn1 = 2 * s + 3; kn1 = kn1 > ktMax ? ktMax : kn1;
    U8 n0 = loadA(kn0);
    U8 n1 = loadA(kn1);

    // stage s complete: only stage s+1's 2 ops may remain outstanding
    asm volatile("s_wait_asynccnt 0x2" ::: "memory");
    __syncthreads();   // publish stage s; proves stage s-1 readers are done

    // issue stage s+2 (real) or dummy into the sink buffer (branch-free)
    {
      int s2 = s + 2;
      int gs = s2 < S ? s2 : (S - 1);
      unsigned la = (s2 < S) ? (lbase + 8192u * (unsigned)(s2 % 3)) : ldummy;
      issueStage(gs, la);
    }

    const unsigned int* sb = &smemB[s % 3][0];
#pragma unroll
    for (int u2 = 0; u2 < 2; ++u2) {
      v16bf a = __builtin_bit_cast(v16bf, u2 ? a1 : a0);
      const unsigned int* sbk = sb + u2 * 1024;
#pragma unroll
      for (int j = 0; j < 4; ++j) {
        const uint4* bt = (const uint4*)(sbk + j * 256 + lane * 8);
        U8 bu;
        bu.lo = bt[0];
        bu.hi = bt[1];
        v16bf b = __builtin_bit_cast(v16bf, bu);
        acc[j] = __builtin_amdgcn_wmma_f32_16x16x32_bf16(
            /*neg_a=*/false, a, /*neg_b=*/false, b,
            /*c_mod=*/(short)0, acc[j], /*reuse_a=*/false, /*reuse_b=*/false);
      }
    }
    a0 = n0;
    a1 = n1;
  }

  // C/D layout: lane 0-15 -> col lane, rows row0+v ; lane 16-31 -> rows +8
  const int rbase = row0 + (hi << 3);
#pragma unroll
  for (int j = 0; j < 4; ++j) {
    const int col = ((mt0 + j) << 4) + (lane & 15);
#pragma unroll
    for (int v = 0; v < 8; ++v) {
      size_t idx = (size_t)(rbase + v) * Mstride + col;
      float val = acc[j][v];
      if (Cres) val += Cres[idx];
      C[idx] = val;
    }
  }
}

// ---------------------------------------------------------------------------
// RMSNorm: one wave per row of [N x H]; bf16 output (or fp32 for final norm)
// ---------------------------------------------------------------------------
__global__ __launch_bounds__(256) void rmsnorm_kernel(
    const float* __restrict__ X, const float* __restrict__ w,
    unsigned short* __restrict__ out_bf, float* __restrict__ out_f32,
    int N, int H)
{
  const int row  = blockIdx.x * 8 + (threadIdx.x >> 5);
  const int lane = threadIdx.x & 31;
  if (row >= N) return;
  const float* x = X + (size_t)row * H;
  float ss = 0.f;
  for (int j = lane; j < H; j += 32) { float v = x[j]; ss += v * v; }
#pragma unroll
  for (int off = 16; off; off >>= 1) ss += __shfl_xor(ss, off, 32);
  const float inv = rsqrtf(ss / (float)H + 1e-6f);
  for (int j = lane; j < H; j += 32) {
    float v = x[j] * inv * w[j];
    if (out_bf) out_bf[(size_t)row * H + j] = f32_to_bf16(v);
    else        out_f32[(size_t)row * H + j] = v;
  }
}

// ---------------------------------------------------------------------------
// mLSTM cell (zero initial state): the matrix memory collapses to the scalar
// s = i_act * dot(q,k) per head; hh = (s/denom)*v; per-head RMSNorm; * sig(og)
// Z row layout (stride 1600): q[0,256) k[256,512) v[512,1024)
//                             i@1024+h  f@1040+h  og[1056,1568)
// One wave per (token, head). Output: bf16 act[N x 512].
// ---------------------------------------------------------------------------
__global__ __launch_bounds__(256) void mlstm_cell_kernel(
    const float* __restrict__ Z,
    const float* __restrict__ bi, const float* __restrict__ bf,
    const float* __restrict__ mh_w,
    unsigned short* __restrict__ act, int N)
{
  const int wid  = blockIdx.x * 8 + (threadIdx.x >> 5);
  const int lane = threadIdx.x & 31;
  if (wid >= N * 8) return;
  const int n = wid >> 3;
  const int h = wid & 7;
  const float* z = Z + (size_t)n * 1600;

  float q = z[h * 32 + lane];
  float k = z[256 + h * 32 + lane];
  float dqk = q * k;
#pragma unroll
  for (int off = 16; off; off >>= 1) dqk += __shfl_xor(dqk, off, 32);

  float i_pre = z[1024 + h] + bi[h];
  float f_pre = z[1040 + h] + bf[h];
  i_pre = 15.f * tanhf(i_pre * (1.f / 15.f));      // soft cap
  f_pre = 15.f * tanhf(f_pre * (1.f / 15.f));
  float f_log = -log1pf(expf(-f_pre));             // log_sigmoid (f_pre in [-15,15])
  float m     = fmaxf(f_log, i_pre);
  float i_act = expf(i_pre - m);
  float s     = i_act * dqk;
  float denom = fmaxf(fabsf(s), expf(-m)) + 1e-6f;
  float coef  = s / denom;

  float v0 = z[512 + h * 64 + lane];
  float v1 = z[512 + h * 64 + 32 + lane];
  float h0 = coef * v0, h1 = coef * v1;
  float ss = h0 * h0 + h1 * h1;
#pragma unroll
  for (int off = 16; off; off >>= 1) ss += __shfl_xor(ss, off, 32);
  float inv = rsqrtf(ss * (1.f / 64.f) + 1e-6f);

  float og0 = z[1056 + h * 64 + lane];
  float og1 = z[1056 + h * 64 + 32 + lane];
  float sg0 = 1.f / (1.f + expf(-og0));
  float sg1 = 1.f / (1.f + expf(-og1));
  float o0 = h0 * inv * mh_w[h * 64 + lane]      * sg0;
  float o1 = h1 * inv * mh_w[h * 64 + 32 + lane] * sg1;
  act[(size_t)n * 512 + h * 64 + lane]      = f32_to_bf16(o0);
  act[(size_t)n * 512 + h * 64 + 32 + lane] = f32_to_bf16(o1);
}

// ---------------------------------------------------------------------------
// SwiGLU: ff[n,j] = silu(G[n,j]) * G[n,1408+j],  G stride 2816 -> bf16 out
// ---------------------------------------------------------------------------
__global__ __launch_bounds__(256) void swiglu_kernel(
    const float* __restrict__ G, unsigned short* __restrict__ ff, int total)
{
  int t = blockIdx.x * 256 + threadIdx.x;
  if (t >= total) return;
  int n = t / 1408, j = t - n * 1408;
  const float* g = G + (size_t)n * 2816;
  float gv = g[j], uv = g[1408 + j];
  float sl = gv / (1.f + expf(-gv));
  ff[t] = f32_to_bf16(sl * uv);
}

// ---------------------------------------------------------------------------
// Host orchestration
// ---------------------------------------------------------------------------
extern "C" void kernel_launch(void* const* d_in, const int* in_sizes, int n_in,
                              void* d_out, int out_size, void* d_ws, size_t ws_size,
                              hipStream_t stream) {
  (void)in_sizes; (void)n_in; (void)out_size; (void)ws_size;
  const float* u       = (const float*)d_in[0];
  const float* norm1_w = (const float*)d_in[1];
  const float* Wq      = (const float*)d_in[2];
  const float* Wk      = (const float*)d_in[3];
  const float* Wv      = (const float*)d_in[4];
  const float* Wi      = (const float*)d_in[5];
  const float* bi      = (const float*)d_in[6];
  const float* Wf      = (const float*)d_in[7];
  const float* bfb     = (const float*)d_in[8];
  const float* mh_w    = (const float*)d_in[9];
  const float* Wog     = (const float*)d_in[10];
  const float* Wout    = (const float*)d_in[11];
  const float* norm2_w = (const float*)d_in[12];
  const float* W_up    = (const float*)d_in[13];
  const float* W_gate  = (const float*)d_in[14];
  const float* W_down  = (const float*)d_in[15];
  const float* final_w = (const float*)d_in[16];
  float* out = (float*)d_out;

  const int N = 2048, H = 512, F = 1408, L = 4;
  const int T_QKV = 100;   // [Wq16|Wk16|Wv32|Wi1|Wf1|Wog32|pad2] -> 1600 cols
  const int T_GU  = 176;   // [W_gate88|W_up88] -> 2816 cols
  const int T_OUT = 32;    // 512 cols
  const int T_DN  = 32;    // 512 cols
  const int KT_H  = H / 32;   // 16
  const int KT_F  = F / 32;   // 44

  uintptr_t p = (uintptr_t)d_ws;
  auto carve = [&](size_t bytes) {
    uintptr_t q = (p + 255) & ~(uintptr_t)255;
    p = q + bytes;
    return (void*)q;
  };

  unsigned* pk_qkv[L]; unsigned* pk_out[L]; unsigned* pk_gu[L]; unsigned* pk_dn[L];
  for (int l = 0; l < L; ++l) pk_qkv[l] = (unsigned*)carve((size_t)T_QKV * KT_H * 256 * 4);
  for (int l = 0; l < L; ++l) pk_out[l] = (unsigned*)carve((size_t)T_OUT * KT_H * 256 * 4);
  for (int l = 0; l < L; ++l) pk_gu[l]  = (unsigned*)carve((size_t)T_GU  * KT_H * 256 * 4);
  for (int l = 0; l < L; ++l) pk_dn[l]  = (unsigned*)carve((size_t)T_DN  * KT_F * 256 * 4);

  float*          x   = (float*)carve((size_t)N * H * 4);
  unsigned short* xn  = (unsigned short*)carve((size_t)N * H * 2);
  float*          Z   = (float*)carve((size_t)N * 2816 * 4);  // shared Z(1600)/G(2816)
  unsigned short* act = (unsigned short*)carve((size_t)N * H * 2);
  unsigned short* ff  = (unsigned short*)carve((size_t)N * F * 2);

  auto repack = [&](const float* W, unsigned* dstBase, int K, int M,
                    int mtOff, int tiles, float scale) {
    int KT = K / 32;
    unsigned* dst = dstBase + (size_t)mtOff * KT * 256;
    int total = tiles * KT * 256;
    repack_b_kernel<<<(total + 255) / 256, 256, 0, stream>>>(W, dst, K, M, tiles, scale);
  };
  auto gemm = [&](const unsigned short* A, const unsigned* Bp, float* C,
                  const float* Cres, int n, int k, int mtiles) {
    int blocks = (n / 128) * (mtiles / 4);   // 8 waves/block, 128 rows x 64 cols
    gemm_bf16_wmma<<<blocks, 256, 0, stream>>>(A, Bp, C, Cres, n, k, mtiles);
  };

  const float kscale = 0.1767766952966369f;  // DHQK^-0.5

  // --- weight repack (bf16 fragments; k-scale folded into Wk) ---
  for (int l = 0; l < L; ++l) {
    repack(Wq   + (size_t)l * H * 256, pk_qkv[l], H, 256,  0, 16, 1.f);
    repack(Wk   + (size_t)l * H * 256, pk_qkv[l], H, 256, 16, 16, kscale);
    repack(Wv   + (size_t)l * H * 512, pk_qkv[l], H, 512, 32, 32, 1.f);
    repack(Wi   + (size_t)l * H * 8,   pk_qkv[l], H,   8, 64,  1, 1.f);
    repack(Wf   + (size_t)l * H * 8,   pk_qkv[l], H,   8, 65,  1, 1.f);
    repack(Wog  + (size_t)l * H * 512, pk_qkv[l], H, 512, 66, 32, 1.f);
    repack(Wog  + (size_t)l * H * 512, pk_qkv[l], H,   0, 98,  2, 0.f);  // zero pad tiles
    repack(Wout + (size_t)l * 512 * H, pk_out[l], 512, H, 0, 32, 1.f);
    repack(W_gate + (size_t)l * H * F, pk_gu[l], H, F,  0, 88, 1.f);
    repack(W_up   + (size_t)l * H * F, pk_gu[l], H, F, 88, 88, 1.f);
    repack(W_down + (size_t)l * F * H, pk_dn[l], F, H,  0, 32, 1.f);
  }

  // --- x = u ---
  hipMemcpyAsync(x, u, (size_t)N * H * sizeof(float), hipMemcpyDeviceToDevice, stream);

  // --- layer stack ---
  for (int l = 0; l < L; ++l) {
    rmsnorm_kernel<<<N / 8, 256, 0, stream>>>(x, norm1_w + l * H, xn, nullptr, N, H);
    gemm(xn, pk_qkv[l], Z, nullptr, N, H, T_QKV);                 // Z [N x 1600]
    mlstm_cell_kernel<<<N, 256, 0, stream>>>(Z, bi + l * 8, bfb + l * 8,
                                             mh_w + (size_t)l * 512, act, N);
    gemm(act, pk_out[l], x, x, N, H, T_OUT);                      // x += act @ Wout
    rmsnorm_kernel<<<N / 8, 256, 0, stream>>>(x, norm2_w + l * H, xn, nullptr, N, H);
    gemm(xn, pk_gu[l], Z, nullptr, N, H, T_GU);                   // G [N x 2816]
    swiglu_kernel<<<(N * F + 255) / 256, 256, 0, stream>>>(Z, ff, N * F);
    gemm(ff, pk_dn[l], x, x, N, F, T_DN);                         // x += ff @ W_down
  }

  // --- final norm -> fp32 output ---
  rmsnorm_kernel<<<N / 8, 256, 0, stream>>>(x, final_w, nullptr, out, N, H);
}
